// FeatureProcessingBlock_40441412059181
// MI455X (gfx1250) — matrix operand order repaired and checked by reading, not compiled, not run on hardware
//
#include <hip/hip_runtime.h>
#include <cstdint>
#include <cstddef>

// ---------------------------------------------------------------------------
// FeatureProcessingBlock on MI455X (gfx1250, wave32, WMMA).
//
// out[b,C',h',w'] = sum_t sum_{c,h,w in window} x[b,c,h,w] *
//                   Ws[t,w,w'] * Hs[t,h,h'] * Cs[t,c,C']
// windowed 64x64; B=4, C=48, H=W=512, T=3.
//
// One workgroup = (b, window p, window q, Cout-group of 16, t).
// Stage A: contract c (K=48->64 padded)   : Y[16cout][64h][64w]   (bf16, LDS)
// Stage B: contract w (K=64)              : ZT[16cout][64w'][64h] (bf16, LDS)
// Stage C: contract h (K=64)              : f32 atomics into global out
// All GEMMs via v_wmma_f32_16x16x32_bf16 (f32 accumulation).
// ---------------------------------------------------------------------------

typedef __attribute__((ext_vector_type(16))) __bf16 bf16x16;
typedef __attribute__((ext_vector_type(2)))  __bf16 bf16x2;
typedef __attribute__((ext_vector_type(8)))  float  f32x8;
typedef __attribute__((ext_vector_type(4)))  uint32_t u32x4;

#define NB   4
#define NC   48
#define NH   512
#define NW   512
#define WSZ  64
#define NT   3

// LDS region sizes in ushort units
#define Y_ELEMS   (16 * 64 * 64)   // 65536
#define ZT_ELEMS  (16 * 64 * 64)   // 65536
#define XS_ELEMS  (256 * 64)       // 16384  (256 spatial x 64 padded c)
#define CST_ELEMS (16 * 64)        // 1024
#define WST_ELEMS (64 * 64)        // 4096
#define HST_ELEMS (64 * 64)        // 4096
#define LDS_USHORTS (Y_ELEMS + ZT_ELEMS + XS_ELEMS + CST_ELEMS + WST_ELEMS + HST_ELEMS)
#define LDS_BYTES   (LDS_USHORTS * 2)  // 313344 < 320KB

// XOR swizzle: row-major [rows][64] ushort, 8-element (16B) chunks XORed by row.
// Keeps 16B alignment for b128 LDS ops when (col % 8 == 0).
__device__ __forceinline__ unsigned swz(unsigned row, unsigned col) {
  unsigned chunk = ((col >> 3) ^ row) & 7u;
  return row * 64u + (chunk << 3) + (col & 7u);
}

// float -> bf16 bits, round-to-nearest-even (scalar fallback path)
__device__ __forceinline__ unsigned short f2bf(float f) {
  unsigned u = __float_as_uint(f);
  u += 0x7fffu + ((u >> 16) & 1u);
  return (unsigned short)(u >> 16);
}

// two floats -> packed bf16 pair (lo in [15:0], hi in [31:16])
__device__ __forceinline__ uint32_t pkbf(float lo, float hi) {
#if __has_builtin(__builtin_amdgcn_cvt_pk_bf16_f32)
  union { bf16x2 v; uint32_t u; } cv;
  cv.v = __builtin_amdgcn_cvt_pk_bf16_f32(lo, hi);
  return cv.u;
#else
  return (uint32_t)f2bf(lo) | ((uint32_t)f2bf(hi) << 16);
#endif
}

union ABUnion {
  u32x4   q[2];
  bf16x16 v;
};

// A-operand (16x32 bf16, M x K): lane m = lane&15 holds row m;
// K chunks {kA..kA+7, kA+16..kA+23}, kA = 8*(lane>>4).
__device__ __forceinline__ bf16x16 loadA(const unsigned short* base,
                                         unsigned mrow0, unsigned k0,
                                         unsigned lane) {
  unsigned m  = mrow0 + (lane & 15u);
  unsigned ka = k0 + ((lane >> 4) << 3);
  ABUnion u;
  u.q[0] = *reinterpret_cast<const u32x4*>(base + swz(m, ka));
  u.q[1] = *reinterpret_cast<const u32x4*>(base + swz(m, ka + 16));
  return u.v;
}

// B-operand (32x16 bf16, K x N) from transposed storage Bt[N][K]:
// lane n = lane&15 holds column n; K chunks {kB..kB+7, kB+16..kB+23}, kB = 8*(lane>>4).
__device__ __forceinline__ bf16x16 loadB(const unsigned short* base,
                                         unsigned nrow0, unsigned k0,
                                         unsigned lane) {
  unsigned n  = nrow0 + (lane & 15u);
  unsigned kb = k0 + ((lane >> 4) << 3);
  ABUnion u;
  u.q[0] = *reinterpret_cast<const u32x4*>(base + swz(n, kb));
  u.q[1] = *reinterpret_cast<const u32x4*>(base + swz(n, kb + 16));
  return u.v;
}

__device__ __forceinline__ f32x8 wmma_bf16(bf16x16 a, bf16x16 b, f32x8 c) {
  return __builtin_amdgcn_wmma_f32_16x16x32_bf16(
      /*neg_a=*/false, a, /*neg_b=*/false, b,
      /*c_mod=*/(short)0, c, /*reuse_a=*/false, /*reuse_b=*/false);
}

// pack 8 f32 accumulator values -> 8 bf16 in a u32x4 (RNE)
__device__ __forceinline__ u32x4 pack8(f32x8 acc) {
  u32x4 pk;
#pragma unroll
  for (unsigned r = 0; r < 4u; ++r)
    pk[r] = pkbf(acc[2u * r], acc[2u * r + 1u]);
  return pk;
}

__global__ void __launch_bounds__(512, 1)
fpb_zero_kernel(float4* __restrict__ p, int n4) {
  int i = blockIdx.x * blockDim.x + threadIdx.x;
  int stride = gridDim.x * blockDim.x;
  for (; i < n4; i += stride) p[i] = make_float4(0.f, 0.f, 0.f, 0.f);
}

__global__ void __launch_bounds__(512, 1)
fpb_kernel(const float* __restrict__ x,   // [B][C][H][W]
           const float* __restrict__ Wsm, // [T][64][64]  (w_in, w_out)
           const float* __restrict__ Hsm, // [T][64][64]  (h_in, h_out)
           const float* __restrict__ Csm, // [T][48][48]  (c_in, c_out)
           float* __restrict__ out)       // [B][C][H][W], pre-zeroed
{
  extern __shared__ __align__(16) unsigned short lds[];
  unsigned short* Y   = lds;
  unsigned short* ZT  = Y   + Y_ELEMS;
  unsigned short* XS  = ZT  + ZT_ELEMS;
  unsigned short* CST = XS  + XS_ELEMS;
  unsigned short* WST = CST + CST_ELEMS;
  unsigned short* HST = WST + WST_ELEMS;

  const unsigned tid  = threadIdx.x;
  const unsigned lane = tid & 31u;
  const unsigned wave = tid >> 5;   // 0..15

  // grid: 4b * 8p * 8q * 3cg * 3t = 2304 workgroups
  unsigned wg = blockIdx.x;
  const unsigned t  = wg % 3u;
  const unsigned cg = (wg / 3u) % 3u;
  const unsigned q  = (wg / 9u) % 8u;
  const unsigned p  = (wg / 72u) % 8u;
  const unsigned b  = wg / 576u;
  const unsigned H0 = p * WSZ;
  const unsigned W0 = q * WSZ;

  // ---- load factor matrices into LDS (transposed, swizzled, bf16) ----
  // CST[j=c_out][c_in]  (Bt of stage A), c padded 48->64 with zeros
  for (unsigned i = tid; i < CST_ELEMS; i += 512u) {
    unsigned j = i >> 6, c = i & 63u;
    float v = (c < NC) ? Csm[t * (NC * NC) + c * NC + cg * 16u + j] : 0.f;
    CST[swz(j, c)] = f2bf(v);
  }
  // WST[w_out][w_in]  (Bt of stage B)
  for (unsigned i = tid; i < WST_ELEMS; i += 512u) {
    unsigned wo = i >> 6, wi = i & 63u;
    WST[swz(wo, wi)] = f2bf(Wsm[t * 4096u + wi * 64u + wo]);
  }
  // HST[h_out][h_in]  (A of stage C)
  for (unsigned i = tid; i < HST_ELEMS; i += 512u) {
    unsigned ho = i >> 6, hi = i & 63u;
    HST[swz(ho, hi)] = f2bf(Hsm[t * 4096u + hi * 64u + ho]);
  }
  // zero the c-padding of XS once (pad slots are never overwritten later)
  for (unsigned i = tid; i < 256u * 16u; i += 512u) {
    unsigned s = i >> 4, c = 48u + (i & 15u);
    XS[swz(s, c)] = 0;
  }
  __syncthreads();

  // =========== Stage A: Y[j][h][w] = sum_c X[c][h][w] * Cs[c][j] ===========
  // GEMM M=4096 (spatial, chunked by 256, 16-tile per wave), K=64 (c padded),
  // N=16 (c_out j). D columns are fixed j with 8 consecutive w -> b128 store.
  const bf16x16 cb0 = loadB(CST, 0, 0,  lane);
  const bf16x16 cb1 = loadB(CST, 0, 32, lane);

  for (unsigned ch = 0; ch < 16u; ++ch) {
    // cooperative stage of X chunk (4 h-rows x 64 w x 48 c), f32 -> bf16,
    // transposed into XS[s][c]; float4 loads along w (coalesced b128)
    for (unsigned i = tid; i < NC * 64u; i += 512u) {
      unsigned c = i >> 6, blk = i & 63u;
      unsigned sl = blk << 2;                       // spatial in chunk, %4==0
      unsigned h = (ch << 2) + (sl >> 6), wc = sl & 63u;
      const float4 v4 = *reinterpret_cast<const float4*>(
          x + (((size_t)b * NC + c) * NH + H0 + h) * NW + W0 + wc);
      uint32_t p01 = pkbf(v4.x, v4.y);
      uint32_t p23 = pkbf(v4.z, v4.w);
      XS[swz(sl + 0u, c)] = (unsigned short)p01;
      XS[swz(sl + 1u, c)] = (unsigned short)(p01 >> 16);
      XS[swz(sl + 2u, c)] = (unsigned short)p23;
      XS[swz(sl + 3u, c)] = (unsigned short)(p23 >> 16);
    }
    __syncthreads();

    // each wave computes one 16-wide spatial tile of this chunk
    const unsigned sl0 = wave << 4;
    bf16x16 a0 = loadA(XS, sl0, 0,  lane);
    bf16x16 a1 = loadA(XS, sl0, 32, lane);
    f32x8 acc = {0.f, 0.f, 0.f, 0.f, 0.f, 0.f, 0.f, 0.f};
    acc = wmma_bf16(a0, cb0, acc);
    acc = wmma_bf16(a1, cb1, acc);

    // D: lane column = c_out j, rows = 8 consecutive spatial (same h row)
    const unsigned j  = lane & 15u;
    const unsigned sg = (ch << 8) + sl0 + ((lane >> 4) << 3);
    const unsigned h  = sg >> 6, w0 = sg & 63u;     // w0 % 8 == 0
    *reinterpret_cast<u32x4*>(Y + swz(j * 64u + h, w0)) = pack8(acc);
    __syncthreads();  // protect XS reuse; final iter publishes Y for stage B
  }

  // ======= Stage B: ZT[j][w'][h] = sum_w Y[j][h][w] * Ws[w][w'] =======
  // per wave: its own c_out channel j; M=h, K=w, N=w'
  const unsigned j = wave;
#pragma unroll
  for (unsigned mt = 0; mt < 4u; ++mt) {
    // A-tiles are invariant across nt: load once per mt
    const bf16x16 aa0 = loadA(Y, j * 64u + mt * 16u, 0,  lane);
    const bf16x16 aa1 = loadA(Y, j * 64u + mt * 16u, 32, lane);
#pragma unroll
    for (unsigned nt = 0; nt < 4u; ++nt) {
      f32x8 acc = {0.f, 0.f, 0.f, 0.f, 0.f, 0.f, 0.f, 0.f};
      acc = wmma_bf16(aa0, loadB(WST, nt * 16u, 0,  lane), acc);
      acc = wmma_bf16(aa1, loadB(WST, nt * 16u, 32, lane), acc);
      // store transposed: ZT row = j*64 + w', col = h (8 contiguous bf16)
      const unsigned wp = nt * 16u + (lane & 15u);
      const unsigned h0 = mt * 16u + ((lane >> 4) << 3);
      *reinterpret_cast<u32x4*>(ZT + swz(j * 64u + wp, h0)) = pack8(acc);
    }
  }
  // no barrier: ZT region of channel j is produced and consumed by wave j only

  // ======= Stage C: out[j][h'][w'] += sum_h Hs[h][h'] * ZT[j][w'][h] =======
  const unsigned cout = cg * 16u + j;
  float* outc = out + (((size_t)b * NC + cout) * NH) * NW;
#pragma unroll
  for (unsigned nt = 0; nt < 4u; ++nt) {
    // B-tiles are invariant across mt: load once per nt
    const bf16x16 bb0 = loadB(ZT, j * 64u + nt * 16u, 0,  lane);
    const bf16x16 bb1 = loadB(ZT, j * 64u + nt * 16u, 32, lane);
    const unsigned wpg = W0 + nt * 16u + (lane & 15u);
#pragma unroll
    for (unsigned mt = 0; mt < 4u; ++mt) {
      f32x8 acc = {0.f, 0.f, 0.f, 0.f, 0.f, 0.f, 0.f, 0.f};
      acc = wmma_bf16(loadA(HST, mt * 16u, 0,  lane), bb0, acc);
      acc = wmma_bf16(loadA(HST, mt * 16u, 32, lane), bb1, acc);
#pragma unroll
      for (unsigned r = 0; r < 8u; ++r) {
        unsigned hpg = H0 + mt * 16u + r + ((lane >> 4) << 3);
        // accumulate over t; relaxed agent-scope fadd -> global_atomic_add_f32
        (void)__hip_atomic_fetch_add(outc + (size_t)hpg * NW + wpg, acc[r],
                                     __ATOMIC_RELAXED, __HIP_MEMORY_SCOPE_AGENT);
      }
    }
  }
}

// ---------------------------------------------------------------------------
extern "C" void kernel_launch(void* const* d_in, const int* in_sizes, int n_in,
                              void* d_out, int out_size, void* d_ws, size_t ws_size,
                              hipStream_t stream) {
  const float* x   = (const float*)d_in[0];
  const float* Wsm = (const float*)d_in[1];
  const float* Hsm = (const float*)d_in[2];
  const float* Csm = (const float*)d_in[3];
  float* out = (float*)d_out;

  // zero output (poisoned by harness; we accumulate over t with atomics)
  int n4 = out_size / 4;
  int zb = (n4 + 511) / 512;
  if (zb > 65535) zb = 65535;
  fpb_zero_kernel<<<zb, 512, 0, stream>>>((float4*)out, n4);

  // idempotent; not a stream op, safe under graph capture
  hipFuncSetAttribute((const void*)fpb_kernel,
                      hipFuncAttributeMaxDynamicSharedMemorySize, LDS_BYTES);

  dim3 grid(NB * 8 * 8 * 3 * NT);  // 2304
  dim3 block(512);
  fpb_kernel<<<grid, block, LDS_BYTES, stream>>>(x, Wsm, Hsm, Csm, out);
}
